// tnn_24060406792344
// MI455X (gfx1250) — matrix operand order repaired and checked by reading, not compiled, not run on hardware
//
#include <hip/hip_runtime.h>
#include <hip/hip_bf16.h>

typedef __attribute__((ext_vector_type(16))) __bf16        v16bf;
typedef __attribute__((ext_vector_type(8)))  float         v8f;
typedef __attribute__((ext_vector_type(4)))  unsigned int  u32x4;
typedef __attribute__((ext_vector_type(4)))  int           v4i;

typedef __attribute__((address_space(1))) v4i as1_v4i;
typedef __attribute__((address_space(3))) v4i as3_v4i;

#define NFEAT   512
#define BATCH   16384
#define NNODES  4095
#define NPAD    4096

#define M_TILE  32
#define N_CHUNK 128
#define THREADS 256

// ---- LDS layout (dynamic shared memory) ----
#define P_OFF   0
#define P_BYTES (M_TILE * NPAD * 2)        // 262144: p values, bf16 [32][4096]
#define A_OFF   (P_OFF + P_BYTES)
#define A_BYTES (M_TILE * NFEAT * 2)       // 32768: A panel bf16 [32][512]
#define B0_OFF  (A_OFF + A_BYTES)          // 294912
#define B_BYTES (N_CHUNK * 32 * 2)         // 8192 per buffer (x3, contiguous)
#define SMEM_BYTES (B0_OFF + 3 * B_BYTES)  // 319488 < 320KB WGP LDS
// after the GEMM phase, A-panel + B-tile regions are reused by the reduce:
#define SL_OFF  A_OFF                      // float[4096] sigmoid(leaves)
#define VH0_OFF (A_OFF + 16384)            // float[4096] v buffers, block-half 0
#define VH1_OFF B0_OFF                     // float[4096] v buffers, block-half 1

// ---- CDNA5 async global->LDS path (ASYNCcnt), with safe fallback ----
#ifndef __has_builtin
#define __has_builtin(x) 0
#endif
#if __has_builtin(__builtin_amdgcn_global_load_async_to_lds_b128) && \
    __has_builtin(__builtin_amdgcn_s_wait_asynccnt)
#define HAVE_ASYNC_LDS 1
#else
#define HAVE_ASYNC_LDS 0
#endif

#if HAVE_ASYNC_LDS
#define ASYNC_WAIT(n) __builtin_amdgcn_s_wait_asynccnt(n)
#else
#define ASYNC_WAIT(n) ((void)0)
#endif

__device__ __forceinline__ void async_copy16(void* lds, const void* g) {
#if HAVE_ASYNC_LDS
    // emits global_load_async_to_lds_b128 (tracked by ASYNCcnt)
    __builtin_amdgcn_global_load_async_to_lds_b128(
        (as1_v4i*)g, (as3_v4i*)lds, 0, 0);
#else
    *(u32x4*)lds = *(const u32x4*)g;
#endif
}

union Frag16 { v16bf v; u32x4 q[2]; };

__device__ __forceinline__ float sigf(float z) {
    // v_exp_f32 + v_add + v_rcp_f32 (avoid the IEEE div expansion)
#if __has_builtin(__builtin_amdgcn_rcpf)
    return __builtin_amdgcn_rcpf(1.0f + __expf(-z));
#else
    return 1.0f / (1.0f + __expf(-z));
#endif
}

// ---- f32 -> bf16 conversion kernels (write workspace) ----
__global__ void cvt_x_kernel(const float* __restrict__ in, __bf16* __restrict__ ob) {
    int i = (blockIdx.x * blockDim.x + threadIdx.x) * 4;
    float4 f = *(const float4*)(in + i);
    ob[i + 0] = (__bf16)f.x;
    ob[i + 1] = (__bf16)f.y;
    ob[i + 2] = (__bf16)f.z;
    ob[i + 3] = (__bf16)f.w;
}

// W is 4095x512; pad to 4096 rows with zeros so GEMM loads stay in-bounds.
__global__ void cvt_w_kernel(const float* __restrict__ in, __bf16* __restrict__ ob) {
    int i = (blockIdx.x * blockDim.x + threadIdx.x) * 4;
    if (i < NNODES * NFEAT) {  // boundary (2096640) is a multiple of 4
        float4 f = *(const float4*)(in + i);
        ob[i + 0] = (__bf16)f.x;
        ob[i + 1] = (__bf16)f.y;
        ob[i + 2] = (__bf16)f.z;
        ob[i + 3] = (__bf16)f.w;
    } else {
        ob[i + 0] = (__bf16)0.0f;
        ob[i + 1] = (__bf16)0.0f;
        ob[i + 2] = (__bf16)0.0f;
        ob[i + 3] = (__bf16)0.0f;
    }
}

// issue one B tile (N_CHUNK nodes x 32 k, bf16): exactly 2 async 16B loads/thread
__device__ __forceinline__ void issue_b_tile(__bf16* buf, const __bf16* __restrict__ wb,
                                             int n0, int k0, int tid) {
    #pragma unroll
    for (int u0 = 0; u0 < (N_CHUNK * 32 * 2) / 16; u0 += THREADS) {
        int u   = u0 + tid;
        int row = u >> 2;
        int c   = (u & 3) * 8;
        async_copy16(buf + row * 32 + c,
                     wb + (size_t)(n0 + row) * NFEAT + k0 + c);
    }
}

__device__ __forceinline__ void epilogue_store(__bf16* pBuf, const v8f& acc, int node, int mrow,
                                               const float* __restrict__ b_all,
                                               const float* __restrict__ s_all) {
    bool valid = node < NNODES;
    float bn = 0.0f, sn = 0.0f;
    if (valid) { bn = b_all[node]; sn = s_all[node]; }
    #pragma unroll
    for (int r = 0; r < 8; ++r) {
        float z = (acc[r] + bn) * sn;
        float p = valid ? sigf(z) : 0.0f;
        pBuf[(mrow + r) * NPAD + node] = (__bf16)p;
    }
}

// ---- fused kernel: bf16 WMMA GEMM + sigmoid + soft-tree reduction ----
__global__ __launch_bounds__(THREADS, 1)
void tree_fused_kernel(const __bf16* __restrict__ xb,
                       const __bf16* __restrict__ wb,
                       const float*  __restrict__ b_all,
                       const float*  __restrict__ s_all,
                       const float*  __restrict__ leaves,
                       float*        __restrict__ out)
{
    extern __shared__ char smem[];
    __bf16* pBuf   = (__bf16*)(smem + P_OFF);
    __bf16* aPanel = (__bf16*)(smem + A_OFF);
    char*   bBase  = smem + B0_OFF;   // 3 contiguous B buffers, stride B_BYTES

    const int tid  = threadIdx.x;
    const int lane = tid & 31;
    const int wave = tid >> 5;
    const int lr   = lane & 15;   // low lane id
    const int hi   = lane >> 4;   // lane half
    const int mi   = wave & 1;         // m-subtile (0..1)
    const int njb  = (wave >> 1) * 2;  // first of this wave's two n-subtiles
    const int mrow = mi * 16 + hi * 8; // base row for C-fragment scatter

    const int rowBase = blockIdx.x * M_TILE;

    // ---- prologue: async-stage A panel + B tiles 0 and 1 ----
    {
        const char* src = (const char*)(xb + (size_t)rowBase * NFEAT);
        char*       dst = (char*)aPanel;
        #pragma unroll
        for (int u0 = 0; u0 < A_BYTES / 16; u0 += THREADS) {
            int u = u0 + tid;
            async_copy16(dst + u * 16, src + u * 16);
        }
    }
    issue_b_tile((__bf16*)(bBase + 0 * B_BYTES), wb, 0, 0, tid);
    issue_b_tile((__bf16*)(bBase + 1 * B_BYTES), wb, 0, 32, tid);
    ASYNC_WAIT(2);          // A panel + tile0 complete; tile1 may still fly
    __syncthreads();

    // ---- pipelined GEMM over 512 tiles: (32 n-chunks) x (16 k-steps) ----
    // triple-buffered: iteration t computes buf[t%3], prefetches tile t+2,
    // and only requires tile t+1 (issued a full iteration ago) at the wait.
    const int NT = (NPAD / N_CHUNK) * (NFEAT / 32);  // 512
    v8f acc0 = {};
    v8f acc1 = {};
    for (int t = 0; t < NT; ++t) {
        const int n0 = (t >> 4) * N_CHUNK;
        const int k0 = (t & 15) * 32;
        const __bf16* cur = (const __bf16*)(bBase + (t % 3) * B_BYTES);

        // prefetch tile t+2 (overlaps with this and next iteration's compute)
        if (t + 2 < NT) {
            const int tn = t + 2;
            issue_b_tile((__bf16*)(bBase + (tn % 3) * B_BYTES),
                         wb, (tn >> 4) * N_CHUNK, (tn & 15) * 32, tid);
        }

        // A fragment: lane lr holds row (mi*16+lr); K = hi*16..+15
        Frag16 fa;
        const __bf16* ap = aPanel + (mi * 16 + lr) * NFEAT + k0 + hi * 16;
        fa.q[0] = *(const u32x4*)(ap);
        fa.q[1] = *(const u32x4*)(ap + 8);

        // B fragments: lane lr holds column n = nj*16+lr; K = hi*16..+15
        Frag16 fb0, fb1;
        const __bf16* bp0 = cur + (njb * 16 + lr) * 32 + hi * 16;
        fb0.q[0] = *(const u32x4*)(bp0);
        fb0.q[1] = *(const u32x4*)(bp0 + 8);
        const __bf16* bp1 = bp0 + 16 * 32;
        fb1.q[0] = *(const u32x4*)(bp1);
        fb1.q[1] = *(const u32x4*)(bp1 + 8);

        acc0 = __builtin_amdgcn_wmma_f32_16x16x32_bf16(
                   false, fa.v, false, fb0.v, (short)0, acc0, false, false);
        acc1 = __builtin_amdgcn_wmma_f32_16x16x32_bf16(
                   false, fa.v, false, fb1.v, (short)0, acc1, false, false);

        // last k-step of this n-chunk: sigmoid epilogue into pBuf (VALU work
        // that also helps cover the async wait below)
        if (k0 == NFEAT - 32) {
            epilogue_store(pBuf, acc0, n0 + (njb + 0) * 16 + lr, mrow, b_all, s_all);
            epilogue_store(pBuf, acc1, n0 + (njb + 1) * 16 + lr, mrow, b_all, s_all);
            acc0 = (v8f){};
            acc1 = (v8f){};
        }

        // require tile t+1 resident; allow tile t+2's 2 loads to remain in flight
        if (t + 2 < NT) ASYNC_WAIT(2);
        else            ASYNC_WAIT(0);
        __syncthreads();
    }

    // ---- tree reduction (bottom-up); two rows in parallel (block halves) ----
    float* sl  = (float*)(smem + SL_OFF);   // sigmoid(leaves), 4096 f32
    const int half = tid >> 7;              // 0 or 1: which row of the pair
    const int ltid = tid & 127;
    float* vh  = (float*)(half ? (smem + VH1_OFF) : (smem + VH0_OFF));
    float* vA  = vh;                        // ping, 2048 f32
    float* vB  = vh + 2048;                 // pong, 2048 f32

    for (int i = tid; i < NPAD; i += THREADS)
        sl[i] = sigf(leaves[i]);
    __syncthreads();

    for (int outer = 0; outer < M_TILE / 2; ++outer) {
        const int r = outer * 2 + half;
        const __bf16* pr = pBuf + r * NPAD;

        // depth 11: nodes [2047..4094], combine sigmoid(leaves) pairs
        for (int i = ltid; i < 2048; i += 128) {
            float p = (float)pr[2047 + i];
            vA[i] = p * sl[2 * i] + (1.0f - p) * sl[2 * i + 1];
        }
        __syncthreads();

        float* src = vA;
        float* dst = vB;
        for (int d = 10; d >= 3; --d) {       // stop at 8 remaining values
            int cnt = 1 << d;
            for (int i = ltid; i < cnt; i += 128) {
                float p = (float)pr[(cnt - 1) + i];
                dst[i] = p * src[2 * i] + (1.0f - p) * src[2 * i + 1];
            }
            __syncthreads();
            float* t = src; src = dst; dst = t;
        }

        // scalar tail: depths 2..0 (8 -> 1), no barriers needed
        if (ltid == 0) {
            float v8v[8];
            #pragma unroll
            for (int j = 0; j < 8; ++j) v8v[j] = src[j];
            float v4v[4];
            #pragma unroll
            for (int i = 0; i < 4; ++i) {
                float p = (float)pr[3 + i];
                v4v[i] = p * v8v[2 * i] + (1.0f - p) * v8v[2 * i + 1];
            }
            float v2v[2];
            #pragma unroll
            for (int i = 0; i < 2; ++i) {
                float p = (float)pr[1 + i];
                v2v[i] = p * v4v[2 * i] + (1.0f - p) * v4v[2 * i + 1];
            }
            float p0 = (float)pr[0];
            out[rowBase + r] = p0 * v2v[0] + (1.0f - p0) * v2v[1];
        }
        __syncthreads();   // buffers reused next iteration
    }
}

extern "C" void kernel_launch(void* const* d_in, const int* in_sizes, int n_in,
                              void* d_out, int out_size, void* d_ws, size_t ws_size,
                              hipStream_t stream) {
    const float* x      = (const float*)d_in[0];
    const float* W      = (const float*)d_in[1];
    const float* b_all  = (const float*)d_in[2];
    const float* s_all  = (const float*)d_in[3];
    const float* leaves = (const float*)d_in[4];
    float*       out    = (float*)d_out;

    __bf16* xb = (__bf16*)d_ws;                                       // 16 MB
    __bf16* wb = (__bf16*)((char*)d_ws + (size_t)BATCH * NFEAT * 2);  // 4 MB

    // f32 -> bf16 conversions (x: 8.4M elems, W: padded 4096x512)
    cvt_x_kernel<<<(BATCH * NFEAT) / (THREADS * 4), THREADS, 0, stream>>>(x, xb);
    cvt_w_kernel<<<(NPAD * NFEAT) / (THREADS * 4), THREADS, 0, stream>>>(W, wb);

    // opt in to >64KB dynamic LDS (WGP has 320KB on gfx1250)
    (void)hipFuncSetAttribute(reinterpret_cast<const void*>(tree_fused_kernel),
                              hipFuncAttributeMaxDynamicSharedMemorySize,
                              SMEM_BYTES);

    tree_fused_kernel<<<BATCH / M_TILE, THREADS, SMEM_BYTES, stream>>>(
        xb, wb, b_all, s_all, leaves, out);
}
